// SimpleTransformerCUDA_47820165874062
// MI455X (gfx1250) — matrix operand order repaired and checked by reading, not compiled, not run on hardware
//
#include <hip/hip_runtime.h>
#include <hip/hip_bf16.h>

// ---------------------------------------------------------------------------
// SimpleTransformer forward on gfx1250 (MI455X), bf16 WMMA path.
//   B=32 S=1024 D=1024 H=16 Hd=64
// v2: GEMM stages A/B block tiles in LDS via global_load_async_to_lds_b128
//     (double-buffered, ASYNCcnt-tracked); attention row-sums via P@ones WMMA.
// ---------------------------------------------------------------------------

#define BATCH    32
#define SEQ      1024
#define DM       1024
#define HEADS    16
#define HDIM     64

typedef __bf16 bf16_t;
typedef __attribute__((ext_vector_type(16))) __bf16 v16bf;
typedef __attribute__((ext_vector_type(8)))  __bf16 v8bf;
typedef __attribute__((ext_vector_type(8)))  float  v8f;

#define LDS_STRIDE 40   // 32 bf16 payload + 8 pad (80B rows, 16B aligned)

// round-to-nearest-even f32 -> bf16
__device__ __forceinline__ bf16_t f2bf(float f) {
  union { float f; unsigned u; } v; v.f = f;
  unsigned r = v.u + 0x7FFFu + ((v.u >> 16) & 1u);
  union { unsigned short s; bf16_t b; } o;
  o.s = (unsigned short)(r >> 16);
  return o.b;
}

__device__ __forceinline__ v8f wmma_bf16(v16bf a, v16bf b, v8f c) {
  return __builtin_amdgcn_wmma_f32_16x16x32_bf16(false, a, false, b,
                                                 (short)0, c, false, false);
}

// A fragment (16x32, MxK): lane l holds row m=l&15; half g=l>>4 selects
// K = {8g..8g+7} in elems 0..7 and K = {16+8g..16+8g+7} in elems 8..15.
template <typename P>
__device__ __forceinline__ v16bf load_a_frag(P base, int ld) {
  const int lane = threadIdx.x & 31;
  const int m = lane & 15;
  const int g = lane >> 4;
  auto row = base + m * ld + 8 * g;
  v8bf lo = *reinterpret_cast<const v8bf*>(&row[0]);
  v8bf hi = *reinterpret_cast<const v8bf*>(&row[16]);
  return __builtin_shufflevector(lo, hi, 0, 1, 2, 3, 4, 5, 6, 7,
                                 8, 9, 10, 11, 12, 13, 14, 15);
}

// B fragment (32x16, KxN) sourced from B^T stored row-major [N][K]:
// lane l holds column n=l&15; elems 0..7 <-> K = 8g+e, elems 8..15 <-> K = 16+8g+(e-8).
template <typename P>
__device__ __forceinline__ v16bf load_bt_frag(P base, int ld) {
  const int lane = threadIdx.x & 31;
  const int n = lane & 15;
  const int g = lane >> 4;
  auto row = base + n * ld + 8 * g;
  v8bf lo = *reinterpret_cast<const v8bf*>(&row[0]);
  v8bf hi = *reinterpret_cast<const v8bf*>(&row[16]);
  return __builtin_shufflevector(lo, hi, 0, 1, 2, 3, 4, 5, 6, 7,
                                 8, 9, 10, 11, 12, 13, 14, 15);
}

__device__ __forceinline__ float hw_max(float v) {
  v = fmaxf(v, __shfl_xor(v, 1, 32));
  v = fmaxf(v, __shfl_xor(v, 2, 32));
  v = fmaxf(v, __shfl_xor(v, 4, 32));
  v = fmaxf(v, __shfl_xor(v, 8, 32));
  return v;
}

// Async copy of one 16B chunk: global -> LDS (ASYNCcnt-tracked).
__device__ __forceinline__ void async_b128(unsigned lds_off, const bf16_t* g) {
  asm volatile("global_load_async_to_lds_b128 %0, %1, off"
               :: "v"(lds_off), "v"(g) : "memory");
}

// Copy a 128x32 bf16 tile (row-major, ldg elements) into LDS (stride
// LDS_STRIDE). 256 threads, 2 chunks each -> +2 ASYNCcnt per wave.
__device__ __forceinline__ void async_cp_tile128x32(const bf16_t* __restrict__ g,
                                                    int ldg, bf16_t* lds,
                                                    int tid) {
#pragma unroll
  for (int c = 0; c < 2; ++c) {
    int chunk = tid + 256 * c;          // 0..511
    int row = chunk >> 2;               // 0..127
    int cc = (chunk & 3) * 8;           // 0,8,16,24 (bf16 elements)
    unsigned l = (unsigned)(uintptr_t)(lds + row * LDS_STRIDE + cc);
    async_b128(l, g + (size_t)row * ldg + cc);
  }
}

// ---------------------------------------------------------------------------
// elementwise helpers
// ---------------------------------------------------------------------------
__global__ __launch_bounds__(256) void cast_f32_bf16(const float* __restrict__ x,
                                                     bf16_t* __restrict__ y) {
  size_t i = (size_t)blockIdx.x * blockDim.x + threadIdx.x;
  y[i] = f2bf(x[i]);
}

__global__ __launch_bounds__(256) void transpose_w(const float* __restrict__ W,
                                                   bf16_t* __restrict__ WT,
                                                   float scale) {
  size_t i = (size_t)blockIdx.x * blockDim.x + threadIdx.x;  // over DM*DM
  int n = (int)(i & (DM - 1));
  int k = (int)(i >> 10);
  WT[(size_t)n * DM + k] = f2bf(W[i] * scale);
}

__global__ __launch_bounds__(256) void transpose_v(const bf16_t* __restrict__ Vb,
                                                   bf16_t* __restrict__ Vt) {
  size_t i = (size_t)blockIdx.x * blockDim.x + threadIdx.x;  // over B*S*DM
  int s = (int)(i & (SEQ - 1));
  int d = (int)((i >> 10) & (HDIM - 1));
  int h = (int)((i >> 16) & (HEADS - 1));
  int b = (int)(i >> 20);
  Vt[i] = Vb[((size_t)(b * SEQ + s)) * DM + h * HDIM + d];
}

// ---------------------------------------------------------------------------
// GEMM: C[M,N] = A[M,K] @ BT[N,K]^T ; bf16 in, fp32 acc.
// Block 256 threads (8 waves 4x2), tile 128x128; K-step 32.
// A/B tiles staged in LDS via async loads, double-buffered.
// ---------------------------------------------------------------------------
template <bool OUT_BF16>
__global__ __launch_bounds__(256)
void gemm_bf16_tn(const bf16_t* __restrict__ A, const bf16_t* __restrict__ BT,
                  void* __restrict__ Cout, int M, int N, int K) {
  __shared__ bf16_t sA[2][128 * LDS_STRIDE];
  __shared__ bf16_t sB[2][128 * LDS_STRIDE];
  const int tid = threadIdx.x;
  const int wave = tid >> 5;
  const int lane = tid & 31;
  const int m0 = blockIdx.x * 128;
  const int n0 = blockIdx.y * 128;
  const int wm = (wave >> 1) * 32;   // wave row offset in block tile
  const int wn = (wave & 1) * 64;    // wave col offset in block tile

  v8f acc[2][4];
#pragma unroll
  for (int i = 0; i < 2; ++i)
#pragma unroll
    for (int j = 0; j < 4; ++j) acc[i][j] = (v8f){0, 0, 0, 0, 0, 0, 0, 0};

  const int nk = K / 32;
  // prologue: prefetch tile 0
  async_cp_tile128x32(A + (size_t)m0 * K, K, &sA[0][0], tid);
  async_cp_tile128x32(BT + (size_t)n0 * K, K, &sB[0][0], tid);

  for (int kt = 0; kt < nk; ++kt) {
    const int buf = kt & 1;
    if (kt + 1 < nk) {
      const int k = (kt + 1) * 32;
      async_cp_tile128x32(A + (size_t)m0 * K + k, K, &sA[buf ^ 1][0], tid);
      async_cp_tile128x32(BT + (size_t)n0 * K + k, K, &sB[buf ^ 1][0], tid);
      // 4 async ops in flight for next tile; current tile (issued earlier,
      // in-order) is complete once count drops to 4.
      asm volatile("s_wait_asynccnt 0x4" ::: "memory");
    } else {
      asm volatile("s_wait_asynccnt 0x0" ::: "memory");
    }
    __syncthreads();

    v16bf a0 = load_a_frag(&sA[buf][(wm + 0) * LDS_STRIDE], LDS_STRIDE);
    v16bf a1 = load_a_frag(&sA[buf][(wm + 16) * LDS_STRIDE], LDS_STRIDE);
    v16bf b[4];
#pragma unroll
    for (int j = 0; j < 4; ++j)
      b[j] = load_bt_frag(&sB[buf][(wn + 16 * j) * LDS_STRIDE], LDS_STRIDE);
#pragma unroll
    for (int j = 0; j < 4; ++j) {
      acc[0][j] = wmma_bf16(a0, b[j], acc[0][j]);
      acc[1][j] = wmma_bf16(a1, b[j], acc[1][j]);
    }
    __syncthreads();  // tile consumed; safe to overwrite next iteration
  }

  const int nl = lane & 15, g = lane >> 4;
#pragma unroll
  for (int i = 0; i < 2; ++i)
#pragma unroll
    for (int j = 0; j < 4; ++j) {
      int col = n0 + wn + 16 * j + nl;
#pragma unroll
      for (int r = 0; r < 8; ++r) {
        size_t idx = (size_t)(m0 + wm + 16 * i + r + 8 * g) * N + col;
        if (OUT_BF16)
          ((bf16_t*)Cout)[idx] = f2bf(acc[i][j][r]);
        else
          ((float*)Cout)[idx] = acc[i][j][r];
      }
    }
}

// ---------------------------------------------------------------------------
// Flash attention: one wave per (b, h, 16-row q-block); k in chunks of 32.
// Row sums computed as P @ ones via WMMA (lands in the same D-layout as
// l_run); row max via 16-lane shfl tree.
// ---------------------------------------------------------------------------
__global__ __launch_bounds__(256)
void attention_fwd(const bf16_t* __restrict__ Q, const bf16_t* __restrict__ Kb,
                   const bf16_t* __restrict__ Vt, bf16_t* __restrict__ ctx) {
  __shared__ bf16_t pbuf[8][16 * LDS_STRIDE];  // per-wave 16x32 P tile
  const int wave = threadIdx.x >> 5;
  const int lane = threadIdx.x & 31;
  const int nl = lane & 15, g = lane >> 4;

  const int wid = blockIdx.x * 8 + wave;      // 0..32767
  const int b = wid >> 10;
  const int h = (wid >> 6) & (HEADS - 1);
  const int qb = wid & 63;

  const bf16_t* Qp = Q + ((size_t)(b * SEQ + qb * 16)) * DM + h * HDIM;
  const bf16_t* Kp = Kb + ((size_t)b * SEQ) * DM + h * HDIM;
  const bf16_t* Vp = Vt + ((size_t)(b * HEADS + h)) * HDIM * SEQ;
  bf16_t* pw = &pbuf[wave][0];

  v16bf qf0 = load_a_frag(Qp, DM);        // d = 0..31
  v16bf qf1 = load_a_frag(Qp + 32, DM);   // d = 32..63

  v16bf ones;
#pragma unroll
  for (int i = 0; i < 16; ++i) ones[i] = (bf16_t)1.0f;

  v8f o[4];
#pragma unroll
  for (int j = 0; j < 4; ++j) o[j] = (v8f){0, 0, 0, 0, 0, 0, 0, 0};
  float m_run[8], l_run[8];
#pragma unroll
  for (int r = 0; r < 8; ++r) { m_run[r] = -1e30f; l_run[r] = 0.0f; }

  const float LOG2E = 1.44269504088896340736f;
  const v8f vzero = (v8f){0, 0, 0, 0, 0, 0, 0, 0};

  for (int kb = 0; kb < SEQ; kb += 32) {
    // scores for columns kb..kb+31 : two 16x16 tiles
    v8f s0 = vzero, s1 = vzero;
    {
      v16bf k0 = load_bt_frag(Kp + (size_t)kb * DM, DM);
      v16bf k1 = load_bt_frag(Kp + (size_t)kb * DM + 32, DM);
      s0 = wmma_bf16(qf0, k0, s0);
      s0 = wmma_bf16(qf1, k1, s0);
      v16bf k2 = load_bt_frag(Kp + (size_t)(kb + 16) * DM, DM);
      v16bf k3 = load_bt_frag(Kp + (size_t)(kb + 16) * DM + 32, DM);
      s1 = wmma_bf16(qf0, k2, s1);
      s1 = wmma_bf16(qf1, k3, s1);
    }
    // online softmax over the 32 new columns
#pragma unroll
    for (int r = 0; r < 8; ++r) {
      float rmax = hw_max(fmaxf(s0[r], s1[r]));
      float m_new = fmaxf(m_run[r], rmax);
      float alpha = __builtin_exp2f((m_run[r] - m_new) * LOG2E);
      float p0 = __builtin_exp2f((s0[r] - m_new) * LOG2E);
      float p1 = __builtin_exp2f((s1[r] - m_new) * LOG2E);
      m_run[r] = m_new;
      l_run[r] *= alpha;
#pragma unroll
      for (int j = 0; j < 4; ++j) o[j][r] *= alpha;
      pw[(r + 8 * g) * LDS_STRIDE + nl] = f2bf(p0);
      pw[(r + 8 * g) * LDS_STRIDE + 16 + nl] = f2bf(p1);
    }
    // P (D-layout in LDS) -> A fragment
    v16bf pf = load_a_frag((const bf16_t*)pw, LDS_STRIDE);
    // row sums of P via matrix op: every column of (P @ ones) is the row sum
    v8f rs = wmma_bf16(pf, ones, vzero);
#pragma unroll
    for (int r = 0; r < 8; ++r) l_run[r] += rs[r];
    // accumulate context
#pragma unroll
    for (int j = 0; j < 4; ++j) {
      v16bf vf = load_bt_frag(Vp + (size_t)(16 * j) * SEQ + kb, SEQ);
      o[j] = wmma_bf16(pf, vf, o[j]);
    }
  }

  bf16_t* cp = ctx + ((size_t)(b * SEQ + qb * 16)) * DM + h * HDIM;
#pragma unroll
  for (int r = 0; r < 8; ++r) {
    float inv = 1.0f / l_run[r];
#pragma unroll
    for (int j = 0; j < 4; ++j)
      cp[(size_t)(r + 8 * g) * DM + 16 * j + nl] = f2bf(o[j][r] * inv);
  }
}

// ---------------------------------------------------------------------------
extern "C" void kernel_launch(void* const* d_in, const int* in_sizes, int n_in,
                              void* d_out, int out_size, void* d_ws, size_t ws_size,
                              hipStream_t stream) {
  const float* x  = (const float*)d_in[0];
  const float* Wq = (const float*)d_in[1];
  const float* Wk = (const float*)d_in[2];
  const float* Wv = (const float*)d_in[3];
  const float* Wo = (const float*)d_in[4];
  float* out = (float*)d_out;

  const size_t BSD = (size_t)BATCH * SEQ * DM;  // 33,554,432 elements
  const size_t WSZ = (size_t)DM * DM;

  char* ws = (char*)d_ws;
  size_t off = 0;
  bf16_t* xb  = (bf16_t*)(ws + off); off += BSD * 2;   // x bf16; reused as ctx
  bf16_t* WqT = (bf16_t*)(ws + off); off += WSZ * 2;
  bf16_t* WkT = (bf16_t*)(ws + off); off += WSZ * 2;
  bf16_t* WvT = (bf16_t*)(ws + off); off += WSZ * 2;
  bf16_t* WoT = (bf16_t*)(ws + off); off += WSZ * 2;
  bf16_t* Qb  = (bf16_t*)(ws + off); off += BSD * 2;
  bf16_t* Kb  = (bf16_t*)(ws + off); off += BSD * 2;
  bf16_t* Vb  = (bf16_t*)(ws + off); off += BSD * 2;
  bf16_t* Vt  = (bf16_t*)(ws + off); off += BSD * 2;   // ~344 MB total

  const int M = BATCH * SEQ;  // 32768

  cast_f32_bf16<<<(int)(BSD / 256), 256, 0, stream>>>(x, xb);
  transpose_w<<<(int)(WSZ / 256), 256, 0, stream>>>(Wq, WqT, 0.125f);  // 1/sqrt(64)
  transpose_w<<<(int)(WSZ / 256), 256, 0, stream>>>(Wk, WkT, 1.0f);
  transpose_w<<<(int)(WSZ / 256), 256, 0, stream>>>(Wv, WvT, 1.0f);
  transpose_w<<<(int)(WSZ / 256), 256, 0, stream>>>(Wo, WoT, 1.0f);

  dim3 ggrid(M / 128, DM / 128);
  gemm_bf16_tn<true><<<ggrid, 256, 0, stream>>>(xb, WqT, Qb, M, DM, DM);
  gemm_bf16_tn<true><<<ggrid, 256, 0, stream>>>(xb, WkT, Kb, M, DM, DM);
  gemm_bf16_tn<true><<<ggrid, 256, 0, stream>>>(xb, WvT, Vb, M, DM, DM);

  transpose_v<<<(int)(BSD / 256), 256, 0, stream>>>(Vb, Vt);

  // xb is dead now; reuse as ctx
  bf16_t* ctx = xb;
  attention_fwd<<<(BATCH * HEADS * (SEQ / 16)) / 8, 256, 0, stream>>>(Qb, Kb, Vt, ctx);

  gemm_bf16_tn<false><<<ggrid, 256, 0, stream>>>(ctx, WoT, out, M, DM, DM);
}
